// Attention_59373627899918
// MI455X (gfx1250) — matrix operand order repaired and checked by reading, not compiled, not run on hardware
//
#include <hip/hip_runtime.h>
#include <math.h>

typedef __attribute__((ext_vector_type(16))) _Float16 v16h;
typedef __attribute__((ext_vector_type(8)))  _Float16 v8h;
typedef __attribute__((ext_vector_type(2)))  _Float16 v2h;
typedef __attribute__((ext_vector_type(8)))  float    v8f;

#define BATCH 8
#define HEADS 16
#define BH    (BATCH*HEADS)
#define SEQ   1024
#define DIM   64
#define ROWS  (BH*SEQ)          // rows per tensor (131072)

// ---------------------------------------------------------------------------
// Pass 1a: L2-normalize q and k rows (D=64) and convert to f16.
// One wave (32 lanes) per row; each lane owns 2 consecutive elements.
// ---------------------------------------------------------------------------
__global__ void norm_qk_f16(const float* __restrict__ q, const float* __restrict__ k,
                            _Float16* __restrict__ qn, _Float16* __restrict__ kn) {
  unsigned gid  = blockIdx.x * blockDim.x + threadIdx.x;
  unsigned wave = gid >> 5;
  unsigned lane = gid & 31;
  const float* src = q;
  _Float16*    dst = qn;
  unsigned row = wave;
  if (wave >= (unsigned)ROWS) { src = k; dst = kn; row = wave - ROWS; }
  float2 x = *((const float2*)(src + (size_t)row * DIM) + lane);
  float ss = x.x * x.x + x.y * x.y;
#pragma unroll
  for (int m = 1; m < 32; m <<= 1) ss += __shfl_xor(ss, m, 32);
  float sc = 1.0f / fmaxf(sqrtf(ss), 1e-12f);
  v2h o;
  o[0] = (_Float16)(x.x * sc);
  o[1] = (_Float16)(x.y * sc);
  *(v2h*)(dst + (size_t)row * DIM + lane * 2) = o;
}

// ---------------------------------------------------------------------------
// Pass 1b: vT[bh][d][n] = (f16) v[bh][n][d]
// Output-coalesced; strided reads are caught by the 192MB L2 (v = 32MB).
// ---------------------------------------------------------------------------
__global__ void v_transpose_f16(const float* __restrict__ v, _Float16* __restrict__ vt) {
  unsigned idx = blockIdx.x * blockDim.x + threadIdx.x;   // bh*65536 + d*1024 + n
  unsigned n  = idx & (SEQ - 1);
  unsigned d  = (idx >> 10) & (DIM - 1);
  unsigned bh = idx >> 16;
  vt[idx] = (_Float16)v[((size_t)bh * SEQ + n) * DIM + d];
}

// ---------------------------------------------------------------------------
// Pass 2: cosine attention; one wave handles TWO 16-row Q tiles (32 q rows)
// so every K/V operand load feeds two WMMA chains (halves L2 traffic/FLOP).
// Per kv-tile (32 rows):
//   S^T(16x16 per subtile, per q-tile) = K_sub(16x64) x Q^T(64x16)  (8 WMMA)
//   P = exp(S^T) -- S^T C-layout maps per-lane onto the P A-layout
//   O(16x64) += P(16x32) x V(32x64) for both q tiles                (8 WMMA)
// No online max: cosine logits bounded by 1 -> exp cannot overflow.
// ---------------------------------------------------------------------------
__global__ void __launch_bounds__(32)
cosattn_wmma(const _Float16* __restrict__ qn, const _Float16* __restrict__ kn,
             const _Float16* __restrict__ vt, float* __restrict__ out) {
  const int wg  = blockIdx.x;
  const int qt  = wg & 31;          // which 32-row q block (32 per head)
  const int bh  = wg >> 5;
  const int b   = bh >> 4;
  const int h   = bh & 15;
  const int lane = threadIdx.x;
  const bool lo  = lane < 16;
  const int q15  = lane & 15;
  const int bsel = lo ? 0 : 16;     // B-matrix contraction offset for lane half
  const int asel = lo ? 0 : 8;      // A-matrix contraction offset for lane half

  const size_t hoff = (size_t)bh * SEQ * DIM;
  const _Float16* Q  = qn + hoff + (size_t)qt * 32 * DIM;
  const _Float16* K  = kn + hoff;
  const _Float16* Vt = vt + hoff;   // [DIM][SEQ]

  // Q^T as B-matrix operands for both q tiles (persistent across kv loop).
  const _Float16* qrowA = Q + q15 * DIM;
  const _Float16* qrowB = qrowA + 16 * DIM;
  const v16h bqa0 = *(const v16h*)(qrowA + bsel);
  const v16h bqa1 = *(const v16h*)(qrowA + 32 + bsel);
  const v16h bqb0 = *(const v16h*)(qrowB + bsel);
  const v16h bqb1 = *(const v16h*)(qrowB + 32 + bsel);

  v8f oa0 = {}, oa1 = {}, oa2 = {}, oa3 = {};
  v8f ob0 = {}, ob1 = {}, ob2 = {}, ob3 = {};
  float rsa = 0.f, rsb = 0.f;       // per-lane partial row sums (q = lane&15)

  for (int kv = 0; kv < SEQ; kv += 32) {
    v8f st0a = {}, st1a = {}, st0b = {}, st1b = {};
    // ---- S^T subtile 0: kv rows [kv, kv+16) ----
    {
      const _Float16* kr = K + (size_t)(kv + q15) * DIM;
      v8h l0 = *(const v8h*)(kr + bsel);
      v8h h0 = *(const v8h*)(kr + 8 + bsel);
      v8h l1 = *(const v8h*)(kr + 32 + bsel);
      v8h h1 = *(const v8h*)(kr + 40 + bsel);
      v16h a0 = __builtin_shufflevector(l0, h0, 0,1,2,3,4,5,6,7,8,9,10,11,12,13,14,15);
      v16h a1 = __builtin_shufflevector(l1, h1, 0,1,2,3,4,5,6,7,8,9,10,11,12,13,14,15);
      st0a = __builtin_amdgcn_wmma_f32_16x16x32_f16(false, a0, false, bqa0, (short)0, st0a, false, false);
      st0a = __builtin_amdgcn_wmma_f32_16x16x32_f16(false, a1, false, bqa1, (short)0, st0a, false, false);
      st0b = __builtin_amdgcn_wmma_f32_16x16x32_f16(false, a0, false, bqb0, (short)0, st0b, false, false);
      st0b = __builtin_amdgcn_wmma_f32_16x16x32_f16(false, a1, false, bqb1, (short)0, st0b, false, false);
    }
    // ---- S^T subtile 1: kv rows [kv+16, kv+32) ----
    {
      const _Float16* kr = K + (size_t)(kv + 16 + q15) * DIM;
      v8h l0 = *(const v8h*)(kr + bsel);
      v8h h0 = *(const v8h*)(kr + 8 + bsel);
      v8h l1 = *(const v8h*)(kr + 32 + bsel);
      v8h h1 = *(const v8h*)(kr + 40 + bsel);
      v16h a0 = __builtin_shufflevector(l0, h0, 0,1,2,3,4,5,6,7,8,9,10,11,12,13,14,15);
      v16h a1 = __builtin_shufflevector(l1, h1, 0,1,2,3,4,5,6,7,8,9,10,11,12,13,14,15);
      st1a = __builtin_amdgcn_wmma_f32_16x16x32_f16(false, a0, false, bqa0, (short)0, st1a, false, false);
      st1a = __builtin_amdgcn_wmma_f32_16x16x32_f16(false, a1, false, bqa1, (short)0, st1a, false, false);
      st1b = __builtin_amdgcn_wmma_f32_16x16x32_f16(false, a0, false, bqb0, (short)0, st1b, false, false);
      st1b = __builtin_amdgcn_wmma_f32_16x16x32_f16(false, a1, false, bqb1, (short)0, st1b, false, false);
    }

    // ---- P = exp(S^T), packed per-lane into WMMA A layout for both q tiles.
    v16h pA, pB;
#pragma unroll
    for (int j = 0; j < 8; ++j) {
      float pa0 = __expf(st0a[j]);
      float pa1 = __expf(st1a[j]);
      float pb0 = __expf(st0b[j]);
      float pb1 = __expf(st1b[j]);
      pA[j] = (_Float16)pa0;  pA[j + 8] = (_Float16)pa1;
      pB[j] = (_Float16)pb0;  pB[j + 8] = (_Float16)pb1;
      rsa += pa0 + pa1;
      rsb += pb0 + pb1;
    }

    // ---- O += P x V; each V chunk load feeds two WMMAs.
    const _Float16* vb = Vt + (size_t)q15 * SEQ + kv + asel;
    {
      v16h v0 = __builtin_shufflevector(*(const v8h*)(vb + (size_t)(0 * 16) * SEQ), *(const v8h*)(vb + (size_t)(0 * 16) * SEQ + 16), 0,1,2,3,4,5,6,7,8,9,10,11,12,13,14,15);
      oa0 = __builtin_amdgcn_wmma_f32_16x16x32_f16(false, pA, false, v0, (short)0, oa0, false, false);
      ob0 = __builtin_amdgcn_wmma_f32_16x16x32_f16(false, pB, false, v0, (short)0, ob0, false, false);
      v16h v1 = __builtin_shufflevector(*(const v8h*)(vb + (size_t)(1 * 16) * SEQ), *(const v8h*)(vb + (size_t)(1 * 16) * SEQ + 16), 0,1,2,3,4,5,6,7,8,9,10,11,12,13,14,15);
      oa1 = __builtin_amdgcn_wmma_f32_16x16x32_f16(false, pA, false, v1, (short)0, oa1, false, false);
      ob1 = __builtin_amdgcn_wmma_f32_16x16x32_f16(false, pB, false, v1, (short)0, ob1, false, false);
      v16h v2 = __builtin_shufflevector(*(const v8h*)(vb + (size_t)(2 * 16) * SEQ), *(const v8h*)(vb + (size_t)(2 * 16) * SEQ + 16), 0,1,2,3,4,5,6,7,8,9,10,11,12,13,14,15);
      oa2 = __builtin_amdgcn_wmma_f32_16x16x32_f16(false, pA, false, v2, (short)0, oa2, false, false);
      ob2 = __builtin_amdgcn_wmma_f32_16x16x32_f16(false, pB, false, v2, (short)0, ob2, false, false);
      v16h v3 = __builtin_shufflevector(*(const v8h*)(vb + (size_t)(3 * 16) * SEQ), *(const v8h*)(vb + (size_t)(3 * 16) * SEQ + 16), 0,1,2,3,4,5,6,7,8,9,10,11,12,13,14,15);
      oa3 = __builtin_amdgcn_wmma_f32_16x16x32_f16(false, pA, false, v3, (short)0, oa3, false, false);
      ob3 = __builtin_amdgcn_wmma_f32_16x16x32_f16(false, pB, false, v3, (short)0, ob3, false, false);
    }
  }

  // Combine lane halves' partial sums (lanes q and q+16 cover disjoint kv).
  float rta = rsa + __shfl_xor(rsa, 16, 32);
  float rtb = rsb + __shfl_xor(rsb, 16, 32);

  // Epilogue: O C-layout row for VGPR j is q = j + asel.
  const size_t row0 = (size_t)b * SEQ + (size_t)qt * 32;
#pragma unroll
  for (int j = 0; j < 8; ++j) {
    int row = j + asel;
    float inva = 1.0f / __shfl(rta, row, 32);
    float invb = 1.0f / __shfl(rtb, row, 32);
    float* opa = out + (row0 + row) * (size_t)(HEADS * DIM) + h * DIM + q15;
    float* opb = opa + (size_t)16 * (HEADS * DIM);
    opa[0]  = oa0[j] * inva;
    opa[16] = oa1[j] * inva;
    opa[32] = oa2[j] * inva;
    opa[48] = oa3[j] * inva;
    opb[0]  = ob0[j] * invb;
    opb[16] = ob1[j] * invb;
    opb[32] = ob2[j] * invb;
    opb[48] = ob3[j] * invb;
  }
}

// ---------------------------------------------------------------------------
extern "C" void kernel_launch(void* const* d_in, const int* in_sizes, int n_in,
                              void* d_out, int out_size, void* d_ws, size_t ws_size,
                              hipStream_t stream) {
  (void)in_sizes; (void)n_in; (void)out_size; (void)ws_size;
  const float* q = (const float*)d_in[0];
  const float* k = (const float*)d_in[1];
  const float* v = (const float*)d_in[2];
  float* out = (float*)d_out;

  _Float16* qn = (_Float16*)d_ws;                       // 16 MB
  _Float16* kn = qn + (size_t)ROWS * DIM;               // 16 MB
  _Float16* vt = kn + (size_t)ROWS * DIM;               // 16 MB

  // Pass 1a: 2*ROWS waves, 32 lanes each -> threads = 2*ROWS*32
  norm_qk_f16<<<(2u * ROWS * 32u) / 256u, 256, 0, stream>>>(q, k, qn, kn);
  // Pass 1b: one thread per element of vT
  v_transpose_f16<<<((unsigned)BH * DIM * SEQ) / 256u, 256, 0, stream>>>(v, vt);
  // Pass 2: one wave per 32-row q block: BH * (SEQ/32) workgroups of 32
  cosattn_wmma<<<BH * (SEQ / 32), 32, 0, stream>>>(qn, kn, vt, out);
}
